// SSIMLoss_24189255811298
// MI455X (gfx1250) — matrix operand order repaired and checked by reading, not compiled, not run on hardware
//
#include <hip/hip_runtime.h>
#include <hip/hip_fp16.h>

typedef __attribute__((ext_vector_type(16))) _Float16 v16h;
typedef __attribute__((ext_vector_type(8)))  _Float16 v8h;
typedef __attribute__((ext_vector_type(8)))  float    v8f;

#define WSZ        11
#define PADR       5
#define IMG_H      256
#define IMG_W      256
#define NPLANES    192                         // 8*8*3
#define ROWS_OUT   16
#define COLS_OUT   128
#define IN_ROWS    26                          // real haloed rows (rest zero)
#define IN_COLS    138                         // real haloed cols (rest zero)
#define LX_ROWS    32                          // padded so A-build needs no row guard
#define LX_PITCH   144                         // padded so A-build needs no col guard (k<=31 at ct=7)
#define HT_ROWS    32                          // h rows (k dimension), 64B per column
#define NQ         5
#define NWAVES     8
#define BLOCKS_PER_PLANE 32                    // (256/16 strips) * (256/128 colblocks)
#define NBLOCKS    (NPLANES * BLOCKS_PER_PLANE)  // 6144
#define NPIX       12582912.0f                 // 192*256*256

// Fused SSIM: separable 11-tap Gaussian conv of {x, y, x2, y2, xy} as
// v_wmma_f32_16x16x32_f16 against a constant banded matrix B[k][n] = w[k-n].
// A-fragments are built with ds_load_b128 from zero-padded f16 LDS tiles and
// v_pk_mul_f16 products (no guards, no f32<->f16 round trips); the H-pass
// result is stored transposed so both the store and the V-pass A-load are
// single b128 LDS ops per fragment-half.
__global__ __launch_bounds__(256) void ssim_fused_kernel(
    const float* __restrict__ pred, const float* __restrict__ tgt,
    float* __restrict__ partial)
{
    __shared__ _Float16 lx[LX_ROWS][LX_PITCH];        // 9216 B
    __shared__ _Float16 ly[LX_ROWS][LX_PITCH];        // 9216 B
    __shared__ _Float16 hqT[NQ][COLS_OUT][HT_ROWS];   // 40960 B, [q][col][k-row]
    __shared__ float    wred[NWAVES];

    const int tid  = threadIdx.x;
    const int lane = tid & 31;           // wave32
    const int wv   = tid >> 5;           // wave id == column tile id (0..7)

    const int wg    = blockIdx.x;
    const int cb    = wg & 1;            // column block (0..1)
    const int strip = (wg >> 1) & 15;    // row strip  (0..15)
    const int plane = wg >> 5;           // 0..191

    const int r0 = strip * ROWS_OUT;
    const int c0 = cb * COLS_OUT;
    const size_t pbase = (size_t)plane * (IMG_H * IMG_W);

    // ---- zero-fill LDS tiles (pad rows/cols stay zero; removes all A-build guards)
    {
        unsigned int* zx = (unsigned int*)&lx[0][0];
        unsigned int* zy = (unsigned int*)&ly[0][0];
        const int nw = (LX_ROWS * LX_PITCH) / 2;   // 2304 dwords per array
        for (int i = tid; i < nw; i += 256) { zx[i] = 0u; zy[i] = 0u; }
    }
    __syncthreads();

    // ---- normalized Gaussian taps (sigma = 1.5), in registers
    float w[WSZ];
    {
        float s = 0.0f;
        #pragma unroll
        for (int t = 0; t < WSZ; ++t) {
            float d = (float)(t - WSZ / 2);
            w[t] = __expf(-(d * d) * (1.0f / (2.0f * 1.5f * 1.5f)));
            s += w[t];
        }
        float inv = 1.0f / (s + 1e-12f);
        #pragma unroll
        for (int t = 0; t < WSZ; ++t) w[t] *= inv;
    }

    // ---- cooperative load of haloed input tile (clamped; OOB stays zero)
    for (int i = tid; i < IN_ROWS * IN_COLS; i += 256) {
        int r = i / IN_COLS, c = i % IN_COLS;
        int gr = r0 - PADR + r, gc = c0 - PADR + c;
        if (gr >= 0 && gr < IMG_H && gc >= 0 && gc < IMG_W) {
            size_t idx = pbase + (size_t)gr * IMG_W + gc;
            lx[r][c] = (_Float16)fminf(fmaxf(pred[idx], 0.0f), 1.0f);
            ly[r][c] = (_Float16)fminf(fmaxf(tgt[idx],  0.0f), 1.0f);
        }
    }
    __syncthreads();

    // ---- constant banded B fragment (32x16, f16): B[k][n] = w[k-n] on band
    // layout (analogous to documented sparse 64x16 B striping):
    // lanes 0-15 hold K=0..7,16..23, lanes 16-31 hold K=8..15,24..31 (matches A); col = lane%16
    v16h bfrag;
    {
        int n = lane & 15;
        #pragma unroll
        for (int e = 0; e < 16; ++e) {
            int k = (e < 8) ? ((lane & 16) >> 1) + e : 8 + ((lane & 16) >> 1) + e;
            int d = k - n;
            bfrag[e] = (_Float16)((d >= 0 && d < WSZ) ? w[d] : 0.0f);
        }
    }

    const v8f zero = {0.f, 0.f, 0.f, 0.f, 0.f, 0.f, 0.f, 0.f};
    const int ct = wv;
    const int m  = lane & 15;
    const int bk = (lane & 16) >> 1;     // A half-fragment K base: 0 or 8

    // ---- horizontal pass: 5 quantities x 2 row groups (rows 0..31, pads are zero)
    #pragma unroll
    for (int rg = 0; rg < 2; ++rg) {
        const int rl = rg * 16 + m;                       // 0..31
        // A(16x32 f16): lane holds k = bk..bk+7 (halfs 0-7) and bk+16..bk+23 (halfs 8-15)
        const _Float16* px = &lx[rl][ct * 16 + bk];
        const _Float16* py = &ly[rl][ct * 16 + bk];
        v8h xl = *(const v8h*)px;
        v8h xh = *(const v8h*)(px + 16);
        v8h yl = *(const v8h*)py;
        v8h yh = *(const v8h*)(py + 16);
        v16h ax = __builtin_shufflevector(xl, xh, 0,1,2,3,4,5,6,7,8,9,10,11,12,13,14,15);
        v16h ay = __builtin_shufflevector(yl, yh, 0,1,2,3,4,5,6,7,8,9,10,11,12,13,14,15);
        v16h axx = ax * ax;      // v_pk_mul_f16; exact same rounding as f32-mul->f16
        v16h ayy = ay * ay;
        v16h axy = ax * ay;

        v8f dx  = __builtin_amdgcn_wmma_f32_16x16x32_f16(false, ax,  false, bfrag, (short)0, zero, false, false);
        v8f dy  = __builtin_amdgcn_wmma_f32_16x16x32_f16(false, ay,  false, bfrag, (short)0, zero, false, false);
        v8f dxx = __builtin_amdgcn_wmma_f32_16x16x32_f16(false, axx, false, bfrag, (short)0, zero, false, false);
        v8f dyy = __builtin_amdgcn_wmma_f32_16x16x32_f16(false, ayy, false, bfrag, (short)0, zero, false, false);
        v8f dxy = __builtin_amdgcn_wmma_f32_16x16x32_f16(false, axy, false, bfrag, (short)0, zero, false, false);

        // D(16x16 f32): lane holds rows M = i + 8*(lane>=16), col N = lane%16.
        // Transposed store: 8 consecutive halfs -> one ds_store_b128 per quantity.
        const int rowbase = rg * 16 + ((lane >> 4) << 3);  // 0/8/16/24 -> 16B aligned
        const int col = ct * 16 + m;
        v8h p0, p1, p2, p3, p4;
        #pragma unroll
        for (int i = 0; i < 8; ++i) {
            p0[i] = (_Float16)dx[i];
            p1[i] = (_Float16)dy[i];
            p2[i] = (_Float16)dxx[i];
            p3[i] = (_Float16)dyy[i];
            p4[i] = (_Float16)dxy[i];
        }
        *(v8h*)&hqT[0][col][rowbase] = p0;
        *(v8h*)&hqT[1][col][rowbase] = p1;
        *(v8h*)&hqT[2][col][rowbase] = p2;
        *(v8h*)&hqT[3][col][rowbase] = p3;
        *(v8h*)&hqT[4][col][rowbase] = p4;
    }
    __syncthreads();

    // ---- vertical pass: A[m][k] = hT[col0+m][k]; contiguous in k -> b128 loads
    v8f acc[NQ];
    #pragma unroll
    for (int q = 0; q < NQ; ++q) {
        const _Float16* ph = &hqT[q][ct * 16 + m][bk];
        v8h hl = *(const v8h*)ph;          // k = bk..bk+7
        v8h hh = *(const v8h*)(ph + 16);   // k = bk+16..bk+23
        v16h a = __builtin_shufflevector(hl, hh, 0,1,2,3,4,5,6,7,8,9,10,11,12,13,14,15);
        acc[q] = __builtin_amdgcn_wmma_f32_16x16x32_f16(false, a, false, bfrag, (short)0, zero, false, false);
    }

    // ---- pointwise SSIM on fragments (all 5 accs share pixel mapping)
    const float C1 = 0.0001f;   // (0.01*1)^2
    const float C2 = 0.0009f;   // (0.03*1)^2
    float lsum = 0.0f;
    #pragma unroll
    for (int i = 0; i < 8; ++i) {
        float mx  = acc[0][i], my  = acc[1][i];
        float exx = acc[2][i], eyy = acc[3][i], exy = acc[4][i];
        float mx2 = mx * mx, my2 = my * my, mxy = mx * my;
        float sx2 = fmaxf(exx - mx2, 0.0f);
        float sy2 = fmaxf(eyy - my2, 0.0f);
        float sxy = exy - mxy;
        float num = (2.0f * mxy + C1) * (2.0f * sxy + C2);
        float den = (mx2 + my2 + C1) * (sx2 + sy2 + C2);
        float v = num / (den + 1e-8f);
        if (!(fabsf(v) < 1e30f)) v = 0.0f;   // nan/inf -> 0 (nan_to_num)
        lsum += v;
    }

    // ---- wave32 + block reduction, deterministic partial per block
    #pragma unroll
    for (int off = 16; off > 0; off >>= 1)
        lsum += __shfl_down(lsum, off, 32);
    if (lane == 0) wred[wv] = lsum;
    __syncthreads();
    if (tid == 0) {
        float s = 0.0f;
        #pragma unroll
        for (int i = 0; i < NWAVES; ++i) s += wred[i];
        partial[blockIdx.x] = s;
    }
}

__global__ __launch_bounds__(256) void ssim_reduce_kernel(
    const float* __restrict__ partial, float* __restrict__ out)
{
    __shared__ float sm[8];
    float s = 0.0f;
    for (int i = threadIdx.x; i < NBLOCKS; i += 256) s += partial[i];
    #pragma unroll
    for (int off = 16; off > 0; off >>= 1)
        s += __shfl_down(s, off, 32);
    const int lane = threadIdx.x & 31, wv = threadIdx.x >> 5;
    if (lane == 0) sm[wv] = s;
    __syncthreads();
    if (threadIdx.x == 0) {
        float t = 0.0f;
        #pragma unroll
        for (int i = 0; i < 8; ++i) t += sm[i];
        out[0] = 1.0f - t / NPIX;
    }
}

extern "C" void kernel_launch(void* const* d_in, const int* in_sizes, int n_in,
                              void* d_out, int out_size, void* d_ws, size_t ws_size,
                              hipStream_t stream) {
    const float* pred = (const float*)d_in[0];
    const float* tgt  = (const float*)d_in[1];
    float* partial = (float*)d_ws;         // NBLOCKS floats, fully rewritten every call
    ssim_fused_kernel<<<NBLOCKS, 256, 0, stream>>>(pred, tgt, partial);
    ssim_reduce_kernel<<<1, 256, 0, stream>>>(partial, (float*)d_out);
}
